// TriangleAttention_86199993630899
// MI455X (gfx1250) — matrix operand order repaired and checked by reading, not compiled, not run on hardware
//
#include <hip/hip_runtime.h>
#include <stdint.h>

typedef __bf16 v16bf __attribute__((ext_vector_type(16)));
typedef __bf16 v8bf  __attribute__((ext_vector_type(8)));
typedef float  v8f   __attribute__((ext_vector_type(8)));

#define WMMA_BF16(a, b, c) \
    __builtin_amdgcn_wmma_f32_16x16x32_bf16(false, (a), false, (b), (short)0, (c), false, false)

// ---------------- geometry ----------------
constexpr int NN   = 256;            // sequence length N
constexpr int DD   = 128;            // channel dim D
constexpr int HH   = 4;              // heads
constexpr int NP   = NN * NN;        // 65536 positions
constexpr float F_SCALE = 0.17677669529663687f; // 1/sqrt(32)

// ---------------- workspace layout (bytes) ----------------
constexpr size_t OFF_WQ   = 0;                       // 384x128 bf16
constexpr size_t OFF_WG   = OFF_WQ   + 384*128*2;    // 128x128 bf16
constexpr size_t OFF_WO   = OFF_WG   + 128*128*2;    // 128x128 bf16
constexpr size_t OFF_WB   = OFF_WO   + 128*128*2;    // 16x128 bf16 (bias weights, rows>=4 zero)
constexpr size_t OFF_Q    = OFF_WB   + 16*128*2;     // (i,h,j,dh) bf16
constexpr size_t OFF_K    = OFF_Q    + (size_t)NP*DD*2;   // (i,h,k,dh) bf16
constexpr size_t OFF_V    = OFF_K    + (size_t)NP*DD*2;   // (i,h,dh,k) bf16  ** transposed **
constexpr size_t OFF_GATE = OFF_V    + (size_t)NP*DD*2;   // (p,f) bf16
constexpr size_t OFF_BIAS = OFF_GATE + (size_t)NP*DD*2;   // (h,p) f32
constexpr size_t OFF_AO   = OFF_BIAS + (size_t)HH*NP*4;   // (p,f) bf16

static __device__ inline float fast_rcp(float x) { return __builtin_amdgcn_rcpf(x); }

// ---------------- fragment helpers (contiguous k order, same for A and B) ----------------
// A 16x32 bf16: lane m=L%16, hi=L/16. elem e -> K = 16*hi+e (contiguous per lane)
static __device__ inline v16bf load_afrag(const __bf16* rowp, int k0, int hi) {
    v16bf a;
#pragma unroll
    for (int e = 0; e < 8; ++e) {
        a[e]     = rowp[k0 + 16*hi + e];
        a[e + 8] = rowp[k0 + 16*hi + 8 + e];
    }
    return a;
}
// B 32x16 bf16: lane n=L%16, hi=L/16. elem e -> K = 16*hi+e (contiguous per lane)
static __device__ inline v16bf load_bfrag(const __bf16* p) {
    return *(const v16bf*)p;   // caller guarantees 32B alignment
}

// ---------------- kernel 0: weights -> bf16 ----------------
__global__ void k_cvt_weights(const float* wqkv, const float* wbias,
                              const float* wgate, const float* wout, char* ws) {
    __bf16* wq = (__bf16*)(ws + OFF_WQ);
    __bf16* wg = (__bf16*)(ws + OFF_WG);
    __bf16* wo = (__bf16*)(ws + OFF_WO);
    __bf16* wb = (__bf16*)(ws + OFF_WB);
    int g = blockIdx.x * 256 + threadIdx.x;          // 328 blocks * 256 = 83968
    if (g < 49152)            wq[g]          = (__bf16)wqkv[g];
    else if (g < 65536)       wg[g - 49152]  = (__bf16)wgate[g - 49152];
    else if (g < 81920)       wo[g - 65536]  = (__bf16)wout[g - 65536];
    else {
        int j = g - 81920;                            // [0, 2048)
        int row = j >> 7, col = j & 127;
        wb[j] = (row < HH) ? (__bf16)wbias[row * 128 + col] : (__bf16)0.f;
    }
}

// ---------------- kernel 1: LayerNorm + fused projections ----------------
// block = 256 threads (8 waves), 16 positions per block, grid = 4096
__global__ __launch_bounds__(256) void k_ln_proj(const float* __restrict__ z,
                          const float* __restrict__ nw, const float* __restrict__ nb,
                          char* ws) {
    const __bf16* wq = (const __bf16*)(ws + OFF_WQ);
    const __bf16* wg = (const __bf16*)(ws + OFF_WG);
    const __bf16* wb = (const __bf16*)(ws + OFF_WB);
    __bf16* qb = (__bf16*)(ws + OFF_Q);
    __bf16* kb = (__bf16*)(ws + OFF_K);
    __bf16* vb = (__bf16*)(ws + OFF_V);
    __bf16* gb = (__bf16*)(ws + OFF_GATE);
    float*  bf = (float*)(ws + OFF_BIAS);

    __shared__ __align__(32) __bf16 Zs[16 * 128];   // 4 KB, LN'd rows (bf16)

    const int tid  = threadIdx.x;
    const int wave = tid >> 5, lane = tid & 31;
    const int hi   = lane >> 4, ln = lane & 15;
    const int pbase = blockIdx.x * 16;
    const int irow  = pbase >> 8;          // uniform: 16-pos tile never crosses a z-row
    const int jbase = pbase & 255;

    // ---- LayerNorm: wave w owns rows 2w, 2w+1; 16 lanes x 8 cols each ----
    {
        const int row = wave * 2 + hi;
        const int c0  = ln * 8;
        const float* zp = z + (size_t)(pbase + row) * 128 + c0;
        float v[8]; float s = 0.f, ss = 0.f;
#pragma unroll
        for (int e = 0; e < 8; ++e) { v[e] = zp[e]; s += v[e]; ss += v[e] * v[e]; }
#pragma unroll
        for (int off = 1; off < 16; off <<= 1) {
            s  += __shfl_xor(s,  off, 32);
            ss += __shfl_xor(ss, off, 32);
        }
        const float mu  = s * (1.f / 128.f);
        const float var = ss * (1.f / 128.f) - mu * mu;
        const float rs  = rsqrtf(var + 1e-5f);
#pragma unroll
        for (int e = 0; e < 8; ++e) {
            const int c = c0 + e;
            Zs[row * 128 + c] = (__bf16)((v[e] - mu) * rs * nw[c] + nb[c]);
        }
    }
    __syncthreads();

    // ---- A fragments for this wave (shared M-tile of 16 rows) ----
    v16bf afr[4];
#pragma unroll
    for (int ks = 0; ks < 4; ++ks) afr[ks] = load_afrag(Zs + ln * 128, ks * 32, hi);

    // ---- 33 column tiles: 24 qkv + 8 gate + 1 bias ----
    for (int t = wave; t < 33; t += 8) {
        const __bf16* wptr; int colbase, kind;
        if (t < 24)      { wptr = wq; colbase = t * 16;        kind = 0; }
        else if (t < 32) { wptr = wg; colbase = (t - 24) * 16; kind = 1; }
        else             { wptr = wb; colbase = 0;             kind = 2; }

        v16bf bfr[4];
#pragma unroll
        for (int ks = 0; ks < 4; ++ks)
            bfr[ks] = load_bfrag(wptr + (size_t)(colbase + ln) * 128 + ks * 32 + hi * 16);
        v8f acc = {0.f, 0.f, 0.f, 0.f, 0.f, 0.f, 0.f, 0.f};
#pragma unroll
        for (int ks = 0; ks < 4; ++ks) acc = WMMA_BF16(afr[ks], bfr[ks], acc);

        if (kind == 0) {
            // sec/h/dh-base uniform per tile (tiles never straddle a head boundary)
            const int sec = colbase >> 7;
            const int cb  = colbase & 127;
            const int h   = cb >> 5;
            const int dh  = (cb & 31) + ln;
            if (sec == 2) {
                // V stored transposed: (i,h,dh,k). Lane's 8 accum rows are 8
                // consecutive k for one dh -> single 16B packed store.
                v8bf pk;
#pragma unroll
                for (int r = 0; r < 8; ++r) pk[r] = (__bf16)acc[r];
                *(v8bf*)(vb + (size_t)(irow * HH + h) * 8192 + (size_t)dh * 256
                             + jbase + hi * 8) = pk;
            } else {
                __bf16* dst = (sec == 0) ? qb : kb;
                const float scale = (sec == 0) ? F_SCALE : 1.f;
                __bf16* base = dst + (size_t)(irow * HH + h) * 8192 + dh;
#pragma unroll
                for (int r = 0; r < 8; ++r)
                    base[(size_t)(jbase + hi * 8 + r) * 32] = (__bf16)(acc[r] * scale);
            }
        } else if (kind == 1) {
#pragma unroll
            for (int r = 0; r < 8; ++r) {
                const int p = pbase + hi * 8 + r;
                gb[(size_t)p * 128 + colbase + ln] =
                    (__bf16)fast_rcp(1.f + __expf(-acc[r]));
            }
        } else if (ln < HH) {
#pragma unroll
            for (int r = 0; r < 8; ++r)
                bf[(size_t)ln * NP + pbase + hi * 8 + r] = acc[r];
        }
    }
}

// ---------------- kernel 2: attention per (i, j-tile, h) ----------------
// block = 128 threads (4 waves); wave owns 16 j-rows; grid = (256, 4, 4)
__global__ __launch_bounds__(128) void k_attn(const unsigned char* __restrict__ mask, char* ws) {
    const __bf16* qb = (const __bf16*)(ws + OFF_Q);
    const __bf16* kb = (const __bf16*)(ws + OFF_K);
    const __bf16* vb = (const __bf16*)(ws + OFF_V);
    const float*  bf = (const float*)(ws + OFF_BIAS);
    __bf16* ao = (__bf16*)(ws + OFF_AO);

    const int i = blockIdx.x, jt = blockIdx.y, h = blockIdx.z;

    __shared__ __align__(32) __bf16 Qs[64 * 32];     //  4 KB
    __shared__ __align__(32) __bf16 Ks[256 * 32];    // 16 KB
    __shared__ __align__(32) __bf16 VTs[32 * 256];   // 16 KB (V^T, copied directly)
    __shared__ __align__(32) __bf16 As[4 * 16 * 256];// 32 KB (attn, per wave)

    const int tid = threadIdx.x;
    const __bf16* qg = qb + (size_t)(i * HH + h) * 8192;
    const __bf16* kg = kb + (size_t)(i * HH + h) * 8192;
    const __bf16* vg = vb + (size_t)(i * HH + h) * 8192;   // already (dh,k)

    for (int idx = tid; idx < 1024; idx += 128)     // K: 256x32 bf16
        ((uint4*)Ks)[idx] = ((const uint4*)kg)[idx];
    for (int idx = tid; idx < 1024; idx += 128)     // V^T: 32x256 bf16, plain copy
        ((uint4*)VTs)[idx] = ((const uint4*)vg)[idx];
    for (int idx = tid; idx < 256; idx += 128)      // Q: 64x32 bf16 (this j-tile)
        ((uint4*)Qs)[idx] = ((const uint4*)(qg + (size_t)jt * 2048))[idx];
    __syncthreads();

    const int wave = tid >> 5, lane = tid & 31;
    const int hi = lane >> 4, ln = lane & 15;
    const int jrowbase = jt * 64 + wave * 16;

    // ---- scores = (Q*scale) K^T : one WMMA per 16-col tile (K-depth 32 = DH) ----
    v16bf aq = load_afrag(Qs + (wave * 16 + ln) * 32, 0, hi);
    v8f sc[16];
#pragma unroll
    for (int tg = 0; tg < 4; ++tg) {                 // groups of 4: clause loads, then WMMAs
        v16bf bk[4];
#pragma unroll
        for (int u = 0; u < 4; ++u)
            bk[u] = load_bfrag(Ks + ((tg * 4 + u) * 16 + ln) * 32 + hi * 16);
        v8f zero = {0.f, 0.f, 0.f, 0.f, 0.f, 0.f, 0.f, 0.f};
#pragma unroll
        for (int u = 0; u < 4; ++u) sc[tg * 4 + u] = WMMA_BF16(aq, bk[u], zero);
    }

    // ---- + bias + mask, row softmax (rows hi*8+r, cols t*16+ln across half-wave) ----
    const float* bp = bf + (size_t)h * NP;
    float rmax[8];
#pragma unroll
    for (int r = 0; r < 8; ++r) rmax[r] = -3.0e38f;
#pragma unroll
    for (int t = 0; t < 16; ++t) {
        const int col = t * 16 + ln;
        const float madd = mask[i * 256 + col] ? 0.f : -3.0e38f;
#pragma unroll
        for (int r = 0; r < 8; ++r) {
            const int jr = jrowbase + hi * 8 + r;
            float s = sc[t][r] + bp[(size_t)jr * 256 + col] + madd;
            sc[t][r] = s;
            rmax[r] = fmaxf(rmax[r], s);
        }
    }
#pragma unroll
    for (int off = 1; off < 16; off <<= 1)
#pragma unroll
        for (int r = 0; r < 8; ++r) rmax[r] = fmaxf(rmax[r], __shfl_xor(rmax[r], off, 32));

    float rsum[8];
#pragma unroll
    for (int r = 0; r < 8; ++r) rsum[r] = 0.f;
#pragma unroll
    for (int t = 0; t < 16; ++t)
#pragma unroll
        for (int r = 0; r < 8; ++r) {
            const float p = __expf(sc[t][r] - rmax[r]);
            sc[t][r] = p;
            rsum[r] += p;
        }
#pragma unroll
    for (int off = 1; off < 16; off <<= 1)
#pragma unroll
        for (int r = 0; r < 8; ++r) rsum[r] += __shfl_xor(rsum[r], off, 32);

    __bf16* Aw = As + wave * 4096;
#pragma unroll
    for (int r = 0; r < 8; ++r) {
        const float rinv = fast_rcp(rsum[r]);
#pragma unroll
        for (int t = 0; t < 16; ++t)
            Aw[(hi * 8 + r) * 256 + t * 16 + ln] = (__bf16)(sc[t][r] * rinv);
    }

    // ---- out = attn @ V : 8 K-steps x 2 N-tiles ----
    v8f oc0 = {0.f, 0.f, 0.f, 0.f, 0.f, 0.f, 0.f, 0.f};
    v8f oc1 = oc0;
#pragma unroll
    for (int kk = 0; kk < 8; ++kk) {
        v16bf aa  = load_afrag(Aw + ln * 256, kk * 32, hi);
        v16bf bv0 = load_bfrag(VTs + (size_t)ln * 256        + kk * 32 + hi * 16);
        v16bf bv1 = load_bfrag(VTs + (size_t)(16 + ln) * 256 + kk * 32 + hi * 16);
        oc0 = WMMA_BF16(aa, bv0, oc0);
        oc1 = WMMA_BF16(aa, bv1, oc1);
    }
#pragma unroll
    for (int r = 0; r < 8; ++r) {
        const int j = jrowbase + hi * 8 + r;
        const size_t p = (size_t)i * 256 + j;
        ao[p * 128 + h * 32 + ln]      = (__bf16)oc0[r];
        ao[p * 128 + h * 32 + 16 + ln] = (__bf16)oc1[r];
    }
}

// ---------------- kernel 3: (gate * attn_out) @ w_out^T -> f32 out ----------------
// block = 256 threads (8 waves), 16 positions, grid = 4096
__global__ __launch_bounds__(256) void k_out(char* ws, float* __restrict__ out) {
    const __bf16* ab = (const __bf16*)(ws + OFF_AO);
    const __bf16* gb = (const __bf16*)(ws + OFF_GATE);
    const __bf16* wo = (const __bf16*)(ws + OFF_WO);

    __shared__ __align__(32) __bf16 Xs[16 * 128];

    const int tid = threadIdx.x;
    const int pbase = blockIdx.x * 16;

    for (int idx = tid; idx < 2048; idx += 256) {
        const size_t g = (size_t)pbase * 128 + idx;
        Xs[idx] = (__bf16)((float)ab[g] * (float)gb[g]);
    }
    __syncthreads();

    const int wave = tid >> 5, lane = tid & 31;
    const int hi = lane >> 4, ln = lane & 15;
    const int colbase = wave * 16;

    v16bf afr[4], bfr[4];
#pragma unroll
    for (int ks = 0; ks < 4; ++ks) {
        afr[ks] = load_afrag(Xs + ln * 128, ks * 32, hi);
        bfr[ks] = load_bfrag(wo + (size_t)(colbase + ln) * 128 + ks * 32 + hi * 16);
    }
    v8f acc = {0.f, 0.f, 0.f, 0.f, 0.f, 0.f, 0.f, 0.f};
#pragma unroll
    for (int ks = 0; ks < 4; ++ks) acc = WMMA_BF16(afr[ks], bfr[ks], acc);

#pragma unroll
    for (int r = 0; r < 8; ++r)
        out[(size_t)(pbase + hi * 8 + r) * 128 + colbase + ln] = acc[r];
}

// ---------------- host launcher ----------------
extern "C" void kernel_launch(void* const* d_in, const int* in_sizes, int n_in,
                              void* d_out, int out_size, void* d_ws, size_t ws_size,
                              hipStream_t stream) {
    const float* z           = (const float*)d_in[0];
    const unsigned char* msk = (const unsigned char*)d_in[1];
    const float* nw          = (const float*)d_in[2];
    const float* nb          = (const float*)d_in[3];
    const float* wqkv        = (const float*)d_in[4];
    const float* wbias       = (const float*)d_in[5];
    const float* wgate       = (const float*)d_in[6];
    const float* wout        = (const float*)d_in[7];
    float* out = (float*)d_out;
    char*  ws  = (char*)d_ws;

    k_cvt_weights<<<328, 256, 0, stream>>>(wqkv, wbias, wgate, wout, ws);
    k_ln_proj<<<NP / 16, 256, 0, stream>>>(z, nw, nb, ws);
    k_attn<<<dim3(NN, 4, HH), 128, 0, stream>>>(msk, ws);
    k_out<<<NP / 16, 256, 0, stream>>>(ws, out);
}
